// EmbedTargetLocSplit_44667659879066
// MI455X (gfx1250) — matrix operand order repaired and checked by reading, not compiled, not run on hardware
//
#include <hip/hip_runtime.h>

typedef __attribute__((ext_vector_type(16))) _Float16 v16h;
typedef __attribute__((ext_vector_type(8)))  float    v8f;

#define B_SZ  8192
#define J_SZ  32
#define D_SZ  256

#define ROWS  128      // batch rows per workgroup
#define NCOLS 128      // output columns per workgroup (half of D)
#define KPAD  264      // padded K stride in halfs (264*2=528 B -> bank rotation)

__global__ __launch_bounds__(256)
void embed_target_loc_split_kernel(const float* __restrict__ input,
                                   const int*   __restrict__ validity,
                                   const float* __restrict__ W1,
                                   const float* __restrict__ b1,
                                   const float* __restrict__ W2,
                                   const float* __restrict__ b2,
                                   float*       __restrict__ out)
{
    __shared__ _Float16 sW2T[NCOLS * KPAD];  // W2 transposed [n][k], f16
    __shared__ _Float16 sH  [ROWS  * KPAD];  // h [row][k], f16
    __shared__ float    sX  [ROWS * 4];      // concat(input, validity)
    __shared__ float    sB2 [NCOLS];

    const int j    = blockIdx.x;          // slot
    const int row0 = blockIdx.y * ROWS;   // batch row base
    const int n0   = blockIdx.z * NCOLS;  // output-column base within D
    const int tid  = threadIdx.x;

    // ---- per-thread column of W1/b1 (d == tid, D_SZ == blockDim.x == 256) ----
    const float w10 = W1[(j*4 + 0)*D_SZ + tid];
    const float w11 = W1[(j*4 + 1)*D_SZ + tid];
    const float w12 = W1[(j*4 + 2)*D_SZ + tid];
    const float w13 = W1[(j*4 + 3)*D_SZ + tid];
    const float b1v = b1[j*D_SZ + tid];

    // ---- stage x: gather [input | validity] for this row block ----
    for (int idx = tid; idx < ROWS*4; idx += 256) {
        const int r = idx >> 2, c = idx & 3;
        const int b = row0 + r;
        float v;
        if (c < 3) v = input[(b*J_SZ + j)*3 + c];
        else       v = (float)validity[b*J_SZ + j];
        sX[idx] = v;
    }
    if (tid < NCOLS) sB2[tid] = b2[j*D_SZ + n0 + tid];

    // ---- W2 half-slot: global fp32 -> f16, stored transposed in LDS ----
    // 128 n-cols x 256 k-rows = 32768 elems; each thread moves 32 float4s.
    #pragma unroll 4
    for (int i = 0; i < 32; ++i) {
        const int lin = (tid + i*256) * 4;       // 0..32764
        const int k   = lin >> 7;                // 0..255
        const int nn  = lin & 127;               // 0..124 step 4
        const float4 v = *(const float4*)&W2[(size_t)(j*D_SZ + k)*D_SZ + n0 + nn];
        sW2T[(nn+0)*KPAD + k] = (_Float16)v.x;
        sW2T[(nn+1)*KPAD + k] = (_Float16)v.y;
        sW2T[(nn+2)*KPAD + k] = (_Float16)v.z;
        sW2T[(nn+3)*KPAD + k] = (_Float16)v.w;
    }
    __syncthreads();

    // ---- stage 1: h = silu(x @ W1 + b1), fp32 VALU, write f16 to LDS ----
    for (int r = 0; r < ROWS; ++r) {
        const float x0 = sX[r*4+0], x1 = sX[r*4+1];
        const float x2 = sX[r*4+2], x3 = sX[r*4+3];
        float h = fmaf(x0, w10, fmaf(x1, w11, fmaf(x2, w12, fmaf(x3, w13, b1v))));
        h = h / (1.0f + __expf(-h));             // silu
        sH[r*KPAD + tid] = (_Float16)h;
    }
    __syncthreads();

    // ---- stage 2: out = h @ W2 + b2 via v_wmma_f32_16x16x32_f16 ----
    const int wave    = tid >> 5;
    const int lane    = tid & 31;
    const int laneRow = lane & 15;   // N (B/C/D frags) or M (A frag)
    const int hiHalf  = lane >> 4;   // which K/M half this lane owns
    const int m0      = wave * 16;   // 8 waves * 16 rows = 128 rows

    v8f acc[8] = {};

    #pragma unroll
    for (int kk = 0; kk < 8; ++kk) {
        const int k0 = kk * 32;
        // A fragment (16x32 f16): lanes 0-15: K {0..7,16..23}; lanes 16-31: {8..15,24..31}
        union { uint4 u[2]; v16h v; } fa;
        const _Float16* pa = &sH[(m0 + laneRow)*KPAD + k0 + hiHalf*8];
        fa.u[0] = *(const uint4*)(pa);
        fa.u[1] = *(const uint4*)(pa + 16);

        #pragma unroll
        for (int nt = 0; nt < 8; ++nt) {
            // B fragment (32x16 f16): lanes 0-15: K {0..7,16..23}; lanes 16-31: {8..15,24..31}
            union { uint4 u[2]; v16h v; } fb;
            const _Float16* pb = &sW2T[(nt*16 + laneRow)*KPAD + k0 + hiHalf*8];
            fb.u[0] = *(const uint4*)(pb);
            fb.u[1] = *(const uint4*)(pb + 16);

            acc[nt] = __builtin_amdgcn_wmma_f32_16x16x32_f16(
                /*neg_a=*/false, fa.v, /*neg_b=*/false, fb.v,
                /*c_mod=*/(short)0, acc[nt],
                /*reuse_a=*/false, /*reuse_b=*/false);
        }
    }

    // ---- epilogue: add b2, store fp32 ----
    // C/D layout: VGPR r -> M = r (lanes 0-15) / M = 8+r (lanes 16-31), N = laneRow
    #pragma unroll
    for (int nt = 0; nt < 8; ++nt) {
        const int   ncol = j*D_SZ + n0 + nt*16 + laneRow;
        const float bias = sB2[nt*16 + laneRow];
        #pragma unroll
        for (int r = 0; r < 8; ++r) {
            const int grow = row0 + m0 + hiHalf*8 + r;
            out[(size_t)grow * (J_SZ*D_SZ) + ncol] = acc[nt][r] + bias;
        }
    }
}

extern "C" void kernel_launch(void* const* d_in, const int* in_sizes, int n_in,
                              void* d_out, int out_size, void* d_ws, size_t ws_size,
                              hipStream_t stream) {
    const float* input    = (const float*)d_in[0];
    const int*   validity = (const int*)  d_in[1];
    const float* W1       = (const float*)d_in[2];
    const float* b1       = (const float*)d_in[3];
    const float* W2       = (const float*)d_in[4];
    const float* b2       = (const float*)d_in[5];
    float*       out      = (float*)d_out;

    dim3 grid(J_SZ, B_SZ / ROWS, D_SZ / NCOLS);   // (32, 64, 2)
    embed_target_loc_split_kernel<<<grid, 256, 0, stream>>>(
        input, validity, W1, b1, W2, b2, out);
}
